// EntropySelector_66949950210526
// MI455X (gfx1250) — compile-verified, matrix operand order checked by reading, not run on hardware
//
#include <hip/hip_runtime.h>
#include <stdint.h>

#ifndef __has_builtin
#define __has_builtin(x) 0
#endif

#define HAS_ASYNC_LDS (__has_builtin(__builtin_amdgcn_global_load_async_to_lds_b128) && \
                       __has_builtin(__builtin_amdgcn_global_store_async_from_lds_b128))

// Problem constants (fixed by the reference / setup_inputs)
#define WSZ        16
#define IMG        384
#define WIN        24            // 384 / 16
#define BIMG       64
#define DFEAT      1536
#define NWIN       (BIMG * WSZ * WSZ)   // 16384 windows total
#define ENT_ELEMS  (BIMG * IMG * IMG)   // 9437184
#define MASK_ELEMS NWIN
#define THRESH     0.299f

typedef _Float16 v16h __attribute__((ext_vector_type(16)));
typedef float    v8f  __attribute__((ext_vector_type(8)));

// Exact pointee type expected by the gfx1250 async-LDS builtins (from clang's
// diagnostic): 'int __attribute__((vector_size(4*sizeof(int))))' in AS1/AS3.
typedef int b128_t __attribute__((vector_size(4 * sizeof(int))));
typedef __attribute__((address_space(1))) b128_t gbl_b128;
typedef __attribute__((address_space(3))) b128_t lds_b128;

// Generic -> AS casts via integer (LDS generic addr low 32 bits == LDS offset per
// CDNA5 aperture rules; global aperture is identity). Avoids cross-AS pointer-cast
// restrictions in clang.
__device__ __forceinline__ gbl_b128* to_gbl(const void* p) {
  return (gbl_b128*)(unsigned long long)p;
}
__device__ __forceinline__ lds_b128* to_lds(const void* p) {
  return (lds_b128*)(unsigned int)(unsigned long long)p;
}

__device__ __forceinline__ void wait_async0() {
#if __has_builtin(__builtin_amdgcn_s_wait_asynccnt)
  __builtin_amdgcn_s_wait_asynccnt(0);
#else
  asm volatile("s_wait_asynccnt 0" ::: "memory");
#endif
}

__device__ __forceinline__ float entropy_of(float x) {
  float p = 1.0f / (1.0f + expf(-x));        // sigmoid (preds ~ N(0,1) -> not in [0,1])
  return -p * logf(fmaxf(p, 1e-5f));
}

// Reduce 3 values per lane across the whole wave (32 lanes) using the matrix unit:
// A(16x32 f16) holds the 96 values (3 per lane, zeros elsewhere), B = ones, so
// D[m][n] = sum_k A[m][k]; summing the 8 accumulator VGPRs gives the half-sum,
// one xor-16 shuffle completes the full wave sum (f32 accumulation throughout).
__device__ __forceinline__ float wave_sum3_wmma(float e0, float e1, float e2) {
  v16h a = {};
  a[0] = (_Float16)e0;
  a[1] = (_Float16)e1;
  a[2] = (_Float16)e2;
  v16h ones;
#pragma unroll
  for (int i = 0; i < 16; ++i) ones[i] = (_Float16)1.0f;
  v8f c = {};
  v8f d = __builtin_amdgcn_wmma_f32_16x16x32_f16(
      /*neg_a=*/false, a, /*neg_b=*/false, ones,
      /*c_mod=*/(short)0, c, /*reuse_a=*/false, /*reuse_b=*/false);
  float s = d[0] + d[1] + d[2] + d[3] + d[4] + d[5] + d[6] + d[7];
  s += __shfl_xor(s, 16, 32);
  return s;
}

// ---------------------------------------------------------------------------
// Phase A: per-window entropy sum -> mask (written directly to its final
// location inside d_out). One block = one 24x24 window; 192 threads x 3 px.
// ---------------------------------------------------------------------------
__global__ __launch_bounds__(192) void entropy_mask_kernel(
    const float* __restrict__ preds, float* __restrict__ mask_out) {
  const int w  = blockIdx.x;              // 0..16383
  const int t  = threadIdx.x;             // 0..191
  const int b  = w >> 8;
  const int wy = (w >> 4) & 15;
  const int wx = w & 15;
  const float* base = preds + ((size_t)b * IMG + (size_t)wy * WIN) * IMG + wx * WIN;

  float e[3];
#pragma unroll
  for (int j = 0; j < 3; ++j) {
    int p = t + 192 * j;                  // 0..575, each pixel exactly once
    int r = p / WIN;
    int c = p - r * WIN;
    e[j] = entropy_of(base[(size_t)r * IMG + c]);
  }

  float s = wave_sum3_wmma(e[0], e[1], e[2]);   // full sum within each wave

  __shared__ float wsum[6];
  if ((t & 31) == 0) wsum[t >> 5] = s;
  __syncthreads();
  if (t == 0) {
    float tot = 0.0f;
#pragma unroll
    for (int i = 0; i < 6; ++i) tot += wsum[i];
    float score = tot * (1.0f / (float)(WIN * WIN));
    mask_out[w] = (score > THRESH) ? 1.0f : 0.0f;
  }
}

// ---------------------------------------------------------------------------
// Phase B: exclusive scan of the 16384 mask bits -> per-window output row k.
// Single block, 512 threads x 32 elements. Scan buffer lives in the entropy
// region of d_out (overwritten later by phase D).
// ---------------------------------------------------------------------------
__global__ __launch_bounds__(512) void scan_kernel(
    const float* __restrict__ mask, uint32_t* __restrict__ scan) {
  __shared__ uint32_t tsum[512];
  const int t = threadIdx.x;

  uint32_t s = 0;
  for (int i = 0; i < 32; ++i) s += (mask[t * 32 + i] > 0.5f) ? 1u : 0u;
  tsum[t] = s;
  __syncthreads();

  for (int off = 1; off < 512; off <<= 1) {
    uint32_t v = (t >= off) ? tsum[t - off] : 0u;
    __syncthreads();
    tsum[t] += v;
    __syncthreads();
  }

  uint32_t excl = tsum[t] - s;           // exclusive prefix for this thread's chunk
  for (int i = 0; i < 32; ++i) {
    uint32_t m = (mask[t * 32 + i] > 0.5f) ? 1u : 0u;
    scan[t * 32 + i] = excl;
    excl += m;
  }
}

// ---------------------------------------------------------------------------
// Phase C: compaction. One block per window (128 threads). Selected windows
// copy their 6KB h_inputs row and the 6KB broadcast input_features row through
// LDS using CDNA5 async global<->LDS b128 ops, and emit (gy,gx) coords.
// ---------------------------------------------------------------------------
__global__ __launch_bounds__(128) void gather_kernel(
    const float* __restrict__ input_features, const float* __restrict__ h_inputs,
    const float* __restrict__ mask, const uint32_t* __restrict__ scan,
    float* __restrict__ out, long long K) {
  const int w = blockIdx.x;
  const int t = threadIdx.x;
  if (mask[w] <= 0.5f) return;           // block-uniform

  const uint32_t k = scan[w];
  const int b = w >> 8;

  const float* srcL = input_features + (size_t)b * DFEAT;
  const float* srcH = h_inputs + (size_t)w * DFEAT;
  float* dstL = out + (size_t)k * DFEAT;
  float* dstH = out + ((size_t)K + k) * DFEAT;

  if (t < 2) {
    float* coords = out + (size_t)2 * (size_t)K * DFEAT + MASK_ELEMS;
    const int n = w & 255;
    coords[(size_t)2 * k + t] = (t == 0) ? (float)(n >> 4) : (float)(n & 15);
  }

#if HAS_ASYNC_LDS
  __shared__ __align__(16) unsigned char lbuf[2 * DFEAT * 4];  // 12288 B
  const int o = t * 16;                   // 128 lanes x 16B x 3 chunks = 6144 B
#pragma unroll
  for (int j = 0; j < 3; ++j) {
    __builtin_amdgcn_global_load_async_to_lds_b128(
        to_gbl((const char*)srcL + o + j * 2048), to_lds(lbuf + o + j * 2048), 0, 0);
    __builtin_amdgcn_global_load_async_to_lds_b128(
        to_gbl((const char*)srcH + o + j * 2048), to_lds(lbuf + 6144 + o + j * 2048), 0, 0);
  }
  wait_async0();                          // each lane re-stores only its own bytes
#pragma unroll
  for (int j = 0; j < 3; ++j) {
    __builtin_amdgcn_global_store_async_from_lds_b128(
        to_gbl((char*)dstL + o + j * 2048), to_lds(lbuf + o + j * 2048), 0, 0);
    __builtin_amdgcn_global_store_async_from_lds_b128(
        to_gbl((char*)dstH + o + j * 2048), to_lds(lbuf + 6144 + o + j * 2048), 0, 0);
  }
  wait_async0();
#else
#pragma unroll
  for (int j = 0; j < 3; ++j) {
    int idx = t + j * 128;                // 384 float4 per 1536-float row
    ((float4*)dstL)[idx] = ((const float4*)srcL)[idx];
    ((float4*)dstH)[idx] = ((const float4*)srcH)[idx];
  }
#endif
}

// ---------------------------------------------------------------------------
// Phase D: write the full entropy map (recompute; cheaper than staging 37.7MB).
// float2-vectorized (entropy region is 8B- but not necessarily 16B-aligned).
// ---------------------------------------------------------------------------
__global__ __launch_bounds__(256) void entropy_write_kernel(
    const float2* __restrict__ preds, float2* __restrict__ ent) {
  size_t i = (size_t)blockIdx.x * 256 + threadIdx.x;   // over 4718592 float2
  float2 x = preds[i];
  float2 r;
  r.x = entropy_of(x.x);
  r.y = entropy_of(x.y);
  ent[i] = r;
}

extern "C" void kernel_launch(void* const* d_in, const int* in_sizes, int n_in,
                              void* d_out, int out_size, void* d_ws, size_t ws_size,
                              hipStream_t stream) {
  (void)in_sizes; (void)n_in; (void)d_ws; (void)ws_size;
  const float* input_features = (const float*)d_in[0];  // [64,1536]
  const float* h_inputs       = (const float*)d_in[1];  // [64,256,1536]
  const float* preds          = (const float*)d_in[2];  // [64,1,384,384]
  float* out = (float*)d_out;

  // out_size = 2*K*1536 + 16384 + 2*K + 9437184  ->  recover K on the host.
  long long K = ((long long)out_size - (long long)MASK_ELEMS - (long long)ENT_ELEMS)
                / (long long)(2 * DFEAT + 2);
  if (K < 0) K = 0;

  float* mask_out = out + (size_t)2 * (size_t)K * DFEAT;            // 16384 floats
  float* ent      = mask_out + MASK_ELEMS + (size_t)2 * (size_t)K;  // 9437184 floats
  uint32_t* scan  = (uint32_t*)ent;  // scratch: overwritten by phase D afterwards

  entropy_mask_kernel<<<NWIN, 192, 0, stream>>>(preds, mask_out);
  scan_kernel<<<1, 512, 0, stream>>>(mask_out, scan);
  gather_kernel<<<NWIN, 128, 0, stream>>>(input_features, h_inputs, mask_out, scan, out, K);
  entropy_write_kernel<<<ENT_ELEMS / 512, 256, 0, stream>>>(
      (const float2*)preds, (float2*)ent);
}
